// SynthesizerAttention_7876970020915
// MI455X (gfx1250) — matrix operand order repaired and checked by reading, not compiled, not run on hardware
//
#include <hip/hip_runtime.h>
#include <hip/hip_bf16.h>

// ---------------------------------------------------------------------------
// Synthesizer attention for MI455X (gfx1250).
// All matmuls: bf16 WMMA (v_wmma_f32_16x16x32_bf16, f32 accum).
// Tiles staged LDS-double-buffered via the Tensor Data Mover
// (tensor_load_to_lds + s_wait_tensorcnt), B operands pre-transposed to NxK
// so both tiles are contiguous-in-K (no in-kernel transpose).
// L=1024 BS=4 D=1024 NH=16 DH=64 H=1024.
// ---------------------------------------------------------------------------

typedef __attribute__((ext_vector_type(16))) __bf16       bf16x16;
typedef __attribute__((ext_vector_type(8)))  float        f32x8;
typedef __attribute__((ext_vector_type(4)))  unsigned int u32x4;
typedef __attribute__((ext_vector_type(8)))  int          i32x8;
typedef __attribute__((ext_vector_type(4)))  int          i32x4;

union Frag16 { uint4 u[2]; __bf16 e[16]; bf16x16 v; };

#if defined(__HIP_DEVICE_COMPILE__) && __has_builtin(__builtin_amdgcn_tensor_load_to_lds)
#define HAVE_TDM 1
#else
#define HAVE_TDM 0
#endif

#if HAVE_TDM
// Issue a 2D TDM load: tile (tileRows x tileK) bf16 elements from a row-major
// tensor (row stride rowStride elems) into LDS at byte offset ldsOff.
// D# bit layout per CDNA5 ISA 8.3/8.4 (group0 128b, group1 256b; 2D => groups
// 2/3 zero / disabled semantics via tile_dim2=0).
__device__ __forceinline__ void tdm_load_2d(unsigned ldsOff, const void* gaddr,
                                            unsigned tileK, unsigned tileRows,
                                            unsigned tensorK, unsigned tensorRows,
                                            unsigned long rowStride)
{
  unsigned long ga = (unsigned long)gaddr;
  u32x4 g0;
  g0[0] = 1u;                                   // count=1, user descriptor
  g0[1] = ldsOff;                               // lds_addr (bytes)
  g0[2] = (unsigned)ga;                         // global_addr[31:0]
  g0[3] = (unsigned)(ga >> 32) | (2u << 30);    // global_addr[56:32] | type=2
  i32x8 g1;
  g1[0] = (int)(1u << 16);                      // data_size=1 (2 bytes), no multicast
  g1[1] = (int)((tensorK & 0xFFFFu) << 16);                         // tensor_dim0[15:0]
  g1[2] = (int)((tensorK >> 16) | ((tensorRows & 0xFFFFu) << 16));  // dim0 hi | dim1 lo
  g1[3] = (int)((tensorRows >> 16) | (tileK << 16));                // dim1 hi | tile_dim0
  g1[4] = (int)(tileRows & 0xFFFFu);                                // tile_dim1 (tile_dim2=0)
  g1[5] = (int)(unsigned)(rowStride & 0xFFFFFFFFul);                // dim0_stride[31:0]
  g1[6] = (int)(unsigned)(rowStride >> 32);                         // dim0_stride[47:32]
  g1[7] = 0;
  i32x4 z4 = {0, 0, 0, 0};
#if defined(__clang_major__) && __clang_major__ >= 23
  i32x8 z8 = {0, 0, 0, 0, 0, 0, 0, 0};
  __builtin_amdgcn_tensor_load_to_lds(g0, g1, z4, z4, z8, 0);
#else
  __builtin_amdgcn_tensor_load_to_lds(g0, g1, z4, z4, 0);
#endif
}
#endif

struct GemmParams {
  const __bf16* A;     // row-major M x K, leading dim lda
  const __bf16* BT;    // row-major N x K (B transposed), leading dim ldbt
  const float*  bias;  // may be null
  const float*  mask;  // row-major M x N additive mask, may be null
  float*        Cf;    // f32 output (or null)
  __bf16*       Cb;    // bf16 output (used when Cf==null)
  int M, N, K;
  int lda, ldbt, ldc;
  int zdiv;            // z -> (zo = z/zdiv, zi = z%zdiv)
  long aOuter, aInner;
  long bOuter, bInner;
  long cOuter, cInner;
  long biasInner;
  int  relu;
};

// 128x128 tile, BK=32. 256 threads = 8 waves; wave grid 4(rows)x2(cols);
// each wave computes 32x64 = 2x4 fragments of 16x16 via 16x16x32 bf16 WMMA.
// LDS double buffered; tiles DMA'd by the TDM (wave 0) and overlapped with
// WMMA on the previous tile.
__launch_bounds__(256)
__global__ void gemm_bf16_wmma(GemmParams p)
{
  constexpr int BM = 128, BN = 128, BK = 32;
  __shared__ __align__(16) __bf16 lsA[2][BM * BK];  // [m][k]
  __shared__ __align__(16) __bf16 lsB[2][BN * BK];  // [n][k]

  const int tid  = threadIdx.x;
  const int lane = tid & 31;
  const int wave = tid >> 5;
  const int wrow = wave & 3;
  const int wcol = wave >> 2;
  const int tileM0 = blockIdx.y * BM;
  const int tileN0 = blockIdx.x * BN;
  const int z  = blockIdx.z;
  const int zo = z / p.zdiv;
  const int zi = z % p.zdiv;

  const __bf16* A  = p.A  + (long)zo * p.aOuter + (long)zi * p.aInner;
  const __bf16* BT = p.BT + (long)zo * p.bOuter + (long)zi * p.bInner;

  f32x8 acc[2][4];
  {
    f32x8 zacc = {};
#pragma unroll
    for (int i = 0; i < 2; ++i)
#pragma unroll
      for (int j = 0; j < 4; ++j) acc[i][j] = zacc;
  }

  auto stage = [&](int kk, int buf) {
#if HAVE_TDM
    if (tid < 32) {  // one wave issues both DMAs (EXEC-independent, wave-level)
      tdm_load_2d((unsigned)(size_t)&lsA[buf][0],
                  A + (long)tileM0 * p.lda + kk,
                  BK, BM, (unsigned)(p.K - kk), (unsigned)(p.M - tileM0),
                  (unsigned long)p.lda);
      tdm_load_2d((unsigned)(size_t)&lsB[buf][0],
                  BT + (long)tileN0 * p.ldbt + kk,
                  BK, BN, (unsigned)(p.K - kk), (unsigned)(p.N - tileN0),
                  (unsigned long)p.ldbt);
    }
#else
    // cooperative fallback: thread loads 16 bf16 of one tile row
    const int r  = tid >> 1;
    const int kc = (tid & 1) * 16;
    uint4 a0 = {}, a1 = {};
    if (tileM0 + r < p.M) {
      const uint4* g = (const uint4*)(A + (long)(tileM0 + r) * p.lda + kk + kc);
      a0 = g[0]; a1 = g[1];
    }
    uint4* sa = (uint4*)&lsA[buf][r * BK + kc];
    sa[0] = a0; sa[1] = a1;
    uint4 b0 = {}, b1 = {};
    if (tileN0 + r < p.N) {
      const uint4* g = (const uint4*)(BT + (long)(tileN0 + r) * p.ldbt + kk + kc);
      b0 = g[0]; b1 = g[1];
    }
    uint4* sb = (uint4*)&lsB[buf][r * BK + kc];
    sb[0] = b0; sb[1] = b1;
#endif
  };
  auto stage_wait = [&]() {
#if HAVE_TDM
    if (tid < 32) __builtin_amdgcn_s_wait_tensorcnt(0);
#endif
  };

  // prologue: fill buffer 0
  stage(0, 0);
  stage_wait();
  __syncthreads();

  int buf = 0;
  for (int kk = 0; kk < p.K; kk += BK) {
    if (kk + BK < p.K) stage(kk + BK, buf ^ 1);  // DMA next tile during compute

    // A fragment (16-bit 16x32): lane<16 -> row=lane,    K in {0..7,16..23}
    //                            lane>=16 -> row=lane-16, K in {8..15,24..31}
    Frag16 af[2], bfr[4];
    const int arow = lane & 15;
    const int akb  = (lane >> 4) * 8;
#pragma unroll
    for (int i = 0; i < 2; ++i) {
      const uint4* sp = (const uint4*)&lsA[buf][(wrow * 32 + i * 16 + arow) * BK + akb];
      af[i].u[0] = sp[0];   // K = akb .. akb+7
      af[i].u[1] = sp[2];   // K = akb+16 .. akb+23
    }
    // B fragment (32x16): lane<16 -> col=lane (K in {0..7,16..23}); lane>=16 -> col=lane-16 (K in {8..15,24..31})
    const int bcol = lane & 15;
    const int bkb  = (lane >> 4) * 8;
#pragma unroll
    for (int j = 0; j < 4; ++j) {
      const uint4* sp = (const uint4*)&lsB[buf][(wcol * 64 + j * 16 + bcol) * BK + bkb];
      bfr[j].u[0] = sp[0]; bfr[j].u[1] = sp[2];
    }

#pragma unroll
    for (int i = 0; i < 2; ++i)
#pragma unroll
      for (int j = 0; j < 4; ++j)
        acc[i][j] = __builtin_amdgcn_wmma_f32_16x16x32_bf16(
            false, af[i].v, false, bfr[j].v, (short)0, acc[i][j], false, false);

    stage_wait();
    __syncthreads();
    buf ^= 1;
  }

  // ---- epilogue: bias (+mask) (+relu), store f32 or bf16 ----
  const float*  bias = p.bias ? p.bias + (long)zi * p.biasInner : nullptr;
  float*  Cf = p.Cf ? p.Cf + (long)zo * p.cOuter + (long)zi * p.cInner : nullptr;
  __bf16* Cb = p.Cb ? p.Cb + (long)zo * p.cOuter + (long)zi * p.cInner : nullptr;
  const int rbase = (lane >> 4) * 8;
  const int ccol  = lane & 15;
#pragma unroll
  for (int i = 0; i < 2; ++i)
#pragma unroll
    for (int j = 0; j < 4; ++j)
#pragma unroll
      for (int r = 0; r < 8; ++r) {
        const int m = tileM0 + wrow * 32 + i * 16 + rbase + r;
        const int n = tileN0 + wcol * 64 + j * 16 + ccol;
        if (m < p.M && n < p.N) {
          float c = acc[i][j][r];
          if (bias)   c += bias[n];
          if (p.mask) c += p.mask[(long)m * p.N + n];
          if (p.relu) c = fmaxf(c, 0.0f);
          if (Cf) Cf[(long)m * p.ldc + n] = c;
          else    Cb[(long)m * p.ldc + n] = (__bf16)c;
        }
      }
}

// row-wise softmax over fixed 1024 f32 cols -> bf16; one block (256 thr) per row,
// one float4 per thread kept in registers across both reductions.
__launch_bounds__(256)
__global__ void softmax_row1024(const float* __restrict__ logits,
                                __bf16* __restrict__ probs)
{
  __shared__ float red[256];
  const int  tid = threadIdx.x;
  const long row = blockIdx.x;
  const float4 vv = ((const float4*)(logits + row * 1024))[tid];

  float m = fmaxf(fmaxf(vv.x, vv.y), fmaxf(vv.z, vv.w));
  red[tid] = m; __syncthreads();
  for (int s = 128; s > 0; s >>= 1) { if (tid < s) red[tid] = fmaxf(red[tid], red[tid + s]); __syncthreads(); }
  m = red[0]; __syncthreads();

  const float e0 = __expf(vv.x - m), e1 = __expf(vv.y - m);
  const float e2 = __expf(vv.z - m), e3 = __expf(vv.w - m);
  red[tid] = (e0 + e1) + (e2 + e3); __syncthreads();
  for (int s = 128; s > 0; s >>= 1) { if (tid < s) red[tid] += red[tid + s]; __syncthreads(); }
  const float inv = 1.0f / red[0];

  union { __bf16 b[4]; uint2 u; } o;
  o.b[0] = (__bf16)(e0 * inv); o.b[1] = (__bf16)(e1 * inv);
  o.b[2] = (__bf16)(e2 * inv); o.b[3] = (__bf16)(e3 * inv);
  ((uint2*)(probs + row * 1024))[tid] = o.u;
}

__launch_bounds__(256)
__global__ void convert_bf16(const float* __restrict__ in, __bf16* __restrict__ out, long n)
{
  for (long i = blockIdx.x * 256L + threadIdx.x; i < n; i += (long)gridDim.x * 256L)
    out[i] = (__bf16)in[i];
}

// Batched transpose: in (z: R x C, row stride ldr, offset zo*inZOuter+zi*inZInner)
// -> out (z: C x R, row stride ldo, offset z*outZ), converted to bf16.
template <typename Tin>
__global__ void transpose_batched(const Tin* __restrict__ in, __bf16* __restrict__ out,
                                  int R, int C, int ldr,
                                  long inZOuter, long inZInner, int zdiv,
                                  long outZ, int ldo)
{
  __shared__ float t[16][17];
  const int z = blockIdx.z;
  const int zo = z / zdiv, zi = z % zdiv;
  const Tin* ip = in + zo * inZOuter + zi * inZInner;
  __bf16*    op = out + (long)z * outZ;
  const int c = blockIdx.x * 16 + threadIdx.x;
  const int r = blockIdx.y * 16 + threadIdx.y;
  if (r < R && c < C) t[threadIdx.y][threadIdx.x] = (float)ip[(long)r * ldr + c];
  __syncthreads();
  const int orow = blockIdx.x * 16 + threadIdx.y;  // = c
  const int ocol = blockIdx.y * 16 + threadIdx.x;  // = r
  if (orow < C && ocol < R) op[(long)orow * ldo + ocol] = (__bf16)t[threadIdx.x][threadIdx.y];
}

// ---------------------------------------------------------------------------
extern "C" void kernel_launch(void* const* d_in, const int* in_sizes, int n_in,
                              void* d_out, int out_size, void* d_ws, size_t ws_size,
                              hipStream_t stream)
{
  (void)in_sizes; (void)n_in; (void)out_size; (void)ws_size;
  constexpr int  L = 1024, BS = 4, D = 1024, NH = 16, DH = 64, H = 1024;
  constexpr long MB_ = 1024L * 1024L;

  const float* x    = (const float*)d_in[0];   // (L,BS,D)
  const float* mask = (const float*)d_in[1];   // (L,L)
  const float* Wi   = (const float*)d_in[2];   // (2D,D)   == B^T for GEMM1
  const float* bi   = (const float*)d_in[3];
  const float* w1   = (const float*)d_in[4];   // (NH,DH,H)
  const float* b1   = (const float*)d_in[5];
  const float* w2   = (const float*)d_in[6];   // (NH,H,L)
  const float* b2   = (const float*)d_in[7];
  const float* Wo   = (const float*)d_in[8];   // (D,D)    == B^T for GEMM5
  const float* bo   = (const float*)d_in[9];
  float* out = (float*)d_out;                  // (L,BS,D)

  char* ws = (char*)d_ws;
  auto alloc = [&](long bytes) -> char* {
    char* p = ws; ws += (bytes + 255) & ~255L; return p;
  };
  __bf16* xb    = (__bf16*)alloc(4096L * 1024 * 2);       // x as (L*BS, D)
  __bf16* Wib   = (__bf16*)alloc(2048L * 1024 * 2);       // Wi  (N=2D x K=D)
  __bf16* w1T   = (__bf16*)alloc(16L * 1024 * 64 * 2);    // per-head (H x DH)
  __bf16* w2T   = (__bf16*)alloc(16L * 1024 * 1024 * 2);  // per-head (L x H)
  __bf16* Wob   = (__bf16*)alloc(1024L * 1024 * 2);       // Wo  (N=D x K=D)
  __bf16* projb = (__bf16*)alloc(4096L * 2048 * 2);       // (L*BS, 2D): [q | v]
  __bf16* vT    = (__bf16*)alloc(64L * 64 * 1024 * 2);    // per (b,h): (DH x L)
  __bf16* hidb  = (__bf16*)alloc(64L * 1024 * 1024 * 2);
  float*  logit = (float*)alloc(64L * 1024 * 1024 * 4);
  __bf16* prob  = (__bf16*)alloc(64L * 1024 * 1024 * 2);
  __bf16* attnb = (__bf16*)alloc(4096L * 1024 * 2);       // (L*BS, D)

  // --- bf16 conversions (Wi/Wo are already NxK) + batched weight transposes ---
  convert_bf16<<<dim3(2048), dim3(256), 0, stream>>>(x,  xb,  4096L * 1024);
  convert_bf16<<<dim3(2048), dim3(256), 0, stream>>>(Wi, Wib, 2048L * 1024);
  convert_bf16<<<dim3(1024), dim3(256), 0, stream>>>(Wo, Wob, 1024L * 1024);
  // w1[h] (DH x H) -> w1T[h] (H x DH)
  transpose_batched<float><<<dim3(64, 4, 16), dim3(16, 16), 0, stream>>>(
      w1, w1T, DH, H, H, 0, (long)DH * H, 16, (long)H * DH, DH);
  // w2[h] (H x L) -> w2T[h] (L x H)
  transpose_batched<float><<<dim3(64, 64, 16), dim3(16, 16), 0, stream>>>(
      w2, w2T, H, L, L, 0, (long)H * L, 16, (long)L * H, H);

  GemmParams p{};

  // GEMM1: proj = x @ Wi^T + bi   (M=4096, N=2048, K=1024) -> bf16
  p = GemmParams{};
  p.A = xb;   p.lda = D;
  p.BT = Wib; p.ldbt = D;
  p.bias = bi; p.biasInner = 0;
  p.Cb = projb; p.ldc = 2 * D;
  p.M = L * BS; p.N = 2 * D; p.K = D;
  p.zdiv = 1;
  gemm_bf16_wmma<<<dim3(16, 32, 1), dim3(256), 0, stream>>>(p);

  // v[b,h] (L tokens x DH) -> vT[z=b*16+h] (DH x L)   (from bf16 proj, v half)
  transpose_batched<__bf16><<<dim3(4, 64, 64), dim3(16, 16), 0, stream>>>(
      projb + D, vT, L, DH, BS * 2 * D, 2 * D, DH, 16, (long)DH * L, L);

  // GEMM2: hid[b,h] = relu(q[b,h] @ w1[h] + b1[h])  (64x: M=L, N=H, K=DH)
  p = GemmParams{};
  p.A = projb; p.lda = BS * 2 * D; p.aOuter = 2 * D; p.aInner = DH;
  p.BT = w1T;  p.ldbt = DH;        p.bOuter = 0;     p.bInner = (long)H * DH;
  p.bias = b1; p.biasInner = H;
  p.Cb = hidb; p.ldc = H; p.cOuter = 16 * MB_; p.cInner = MB_;
  p.M = L; p.N = H; p.K = DH;
  p.zdiv = NH; p.relu = 1;
  gemm_bf16_wmma<<<dim3(8, 8, 64), dim3(256), 0, stream>>>(p);

  // GEMM3: logits[b,h] = hid[b,h] @ w2[h] + b2[h] + mask  (64x: M=L, N=L, K=H)
  p = GemmParams{};
  p.A = hidb;  p.lda = H; p.aOuter = 16 * MB_; p.aInner = MB_;
  p.BT = w2T;  p.ldbt = H; p.bOuter = 0;       p.bInner = (long)L * H;
  p.bias = b2; p.biasInner = L;
  p.mask = mask;
  p.Cf = logit; p.ldc = L; p.cOuter = 16 * MB_; p.cInner = MB_;
  p.M = L; p.N = L; p.K = H;
  p.zdiv = NH;
  gemm_bf16_wmma<<<dim3(8, 8, 64), dim3(256), 0, stream>>>(p);

  // softmax over 64*1024 rows of 1024 -> bf16 probs
  softmax_row1024<<<dim3(64 * 1024), dim3(256), 0, stream>>>(logit, prob);

  // GEMM4: attn[b,h] = probs[b,h] @ v[b,h]  (64x: M=L, N=DH, K=L)
  p = GemmParams{};
  p.A = prob; p.lda = L; p.aOuter = 16 * MB_; p.aInner = MB_;
  p.BT = vT;  p.ldbt = L; p.bOuter = 16L * DH * L; p.bInner = (long)DH * L;
  p.Cb = attnb; p.ldc = D; p.cOuter = D; p.cInner = DH;  // attn[(l+b), h*DH+d]
  p.M = L; p.N = DH; p.K = L;
  p.zdiv = NH;
  gemm_bf16_wmma<<<dim3(1, 8, 64), dim3(256), 0, stream>>>(p);

  // GEMM5: out = attn @ Wo^T + bo  (M=4096, N=1024, K=1024) -> f32 to d_out
  p = GemmParams{};
  p.A = attnb; p.lda = D;
  p.BT = Wob;  p.ldbt = D;
  p.bias = bo; p.biasInner = 0;
  p.Cf = out;  p.ldc = D;
  p.M = L * BS; p.N = D; p.K = D;
  p.zdiv = 1;
  gemm_bf16_wmma<<<dim3(8, 32, 1), dim3(256), 0, stream>>>(p);
}
